// HeterogeneousGraphTransformerWithCrossAttention_71588514890090
// MI455X (gfx1250) — matrix operand order repaired and checked by reading, not compiled, not run on hardware
//
#include <hip/hip_runtime.h>
#include <hip/hip_bf16.h>
#include <math.h>

// ---------------------------------------------------------------------------
// MI455X (gfx1250) heterogeneous graph transformer.
// All dense math -> v_wmma_f32_16x16x32_bf16 (wave32 WMMA).
// Fragments load as pairs of ds_load_b128 (16 contiguous bf16), B-operands
// stored transposed in LDS, rows padded (40/72 bf16) for conflict-free b128.
// GEMMs are LDS ping-pong double-buffered (1 barrier / k-step, global fetch
// of tile i+1 overlaps WMMA of tile i) and templated on epilogue so the hot
// loop is branch-free.
// ---------------------------------------------------------------------------

static constexpr int NH = 4;    // heads
static constexpr int DH = 64;   // head dim
static constexpr int HD = 256;  // hidden = NH*DH
static constexpr int OD = 128;  // output dim
static constexpr float SCALE = 0.125f;  // 1/sqrt(DH)

typedef __attribute__((ext_vector_type(16))) __bf16 v16bf;
typedef __attribute__((ext_vector_type(8)))  __bf16 v8bf;
typedef __attribute__((ext_vector_type(4)))  __bf16 v4bf;
typedef __attribute__((ext_vector_type(8)))  float  v8f;

enum { EP_BIAS = 0, EP_ADD = 1, EP_GATE = 2 };

__device__ __forceinline__ v16bf cat8(v8bf lo, v8bf hi) {
  return __builtin_shufflevector(lo, hi, 0, 1, 2, 3, 4, 5, 6, 7,
                                         8, 9, 10, 11, 12, 13, 14, 15);
}

__device__ __forceinline__ float gelu_f(float x) {
  return 0.5f * x * (1.0f + erff(x * 0.70710678118654752440f));
}

__device__ __forceinline__ void atomic_max_f(float* addr, float v) {
  // monotone-in-float trick: positive -> int max, negative -> uint min.
  if (v >= 0.0f) atomicMax((int*)addr, __float_as_int(v));
  else           atomicMin((unsigned int*)addr, __float_as_uint(v));
}

// Stage one 64x32 A tile (fp32 -> bf16, optional GELU) and one 32x64 W tile
// (transposed into [n][k]) into LDS.
template <int GELUA>
__device__ __forceinline__ void stage_tiles(const float* __restrict__ A,
                                            const float* __restrict__ W,
                                            int mb, int nb, int K, int Nn, int k0,
                                            int tid, __bf16 (*as)[40], __bf16 (*bt)[40])
{
  for (int i = tid; i < 512; i += 256) {
    int r = i >> 3, c4 = (i & 7) * 4;
    float4 f = *(const float4*)&A[(size_t)(mb + r) * K + k0 + c4];
    if (GELUA) { f.x = gelu_f(f.x); f.y = gelu_f(f.y); f.z = gelu_f(f.z); f.w = gelu_f(f.w); }
    v4bf p = { (__bf16)f.x, (__bf16)f.y, (__bf16)f.z, (__bf16)f.w };
    *(v4bf*)&as[r][c4] = p;
  }
  for (int i = tid; i < 512; i += 256) {
    int r = i >> 4, c4 = (i & 15) * 4;
    float4 f = *(const float4*)&W[(size_t)(k0 + r) * Nn + nb + c4];
    bt[c4 + 0][r] = (__bf16)f.x; bt[c4 + 1][r] = (__bf16)f.y;
    bt[c4 + 2][r] = (__bf16)f.z; bt[c4 + 3][r] = (__bf16)f.w;
  }
}

// ---------------------------------------------------------------------------
// Generic GEMM: C[M,Nn] = epilogue( act(A)[M,K] @ W[K,Nn] + bias[Nn] )
//   EP_BIAS: C = AW+b      EP_ADD: C += AW+b
//   EP_GATE: C = g*(AW+b) + (1-g)*R,  g = sigmoid(*gatep); GELUA gelus A.
// Block: 64x64 tile, 8 waves, each wave owns 16x32 (2 C frags). Ping-pong LDS.
// ---------------------------------------------------------------------------
template <int MODE, int GELUA>
__global__ __launch_bounds__(256)
void gemm_bf16_wmma(const float* __restrict__ A, const float* __restrict__ W,
                    const float* __restrict__ bias, float* C,
                    const float* __restrict__ R, const float* __restrict__ gatep,
                    int M, int K, int Nn)
{
  __shared__ __bf16 As[2][64][40];   // [m][k]
  __shared__ __bf16 Bt[2][64][40];   // [n][k]  (W transposed into LDS)
  const int tid  = threadIdx.x;
  const int wave = tid >> 5, lane = tid & 31;
  const int mb = blockIdx.x * 64, nb = blockIdx.y * 64;
  const int wm = (wave >> 1) * 16, wn = (wave & 1) * 32;
  const int m = lane & 15, sel = lane >> 4;
  v8f c0 = {}, c1 = {};
  const int nIter = K >> 5;
  stage_tiles<GELUA>(A, W, mb, nb, K, Nn, 0, tid, As[0], Bt[0]);
  __syncthreads();
  for (int it = 0; it < nIter; ++it) {
    const int p = it & 1;
    if (it + 1 < nIter)
      stage_tiles<GELUA>(A, W, mb, nb, K, Nn, (it + 1) << 5, tid, As[p ^ 1], Bt[p ^ 1]);
    const v8bf* ar  = (const v8bf*)&As[p][wm + m][0];
    const v8bf* br0 = (const v8bf*)&Bt[p][wn + m][0];
    const v8bf* br1 = (const v8bf*)&Bt[p][wn + 16 + m][0];
    v16bf a  = cat8(ar[2 * sel], ar[2 * sel + 1]);
    v16bf b0 = cat8(br0[2 * sel], br0[2 * sel + 1]);
    v16bf b1 = cat8(br1[2 * sel], br1[2 * sel + 1]);
    c0 = __builtin_amdgcn_wmma_f32_16x16x32_bf16(false, a, false, b0, (short)0, c0, false, false);
    c1 = __builtin_amdgcn_wmma_f32_16x16x32_bf16(false, a, false, b1, (short)0, c1, false, false);
    __syncthreads();
  }
  float g = 0.0f;
  if (MODE == EP_GATE) g = 1.0f / (1.0f + __expf(-gatep[0]));
#pragma unroll
  for (int v = 0; v < 8; ++v) {
    int row  = mb + wm + v + 8 * sel;
    int col0 = nb + wn + m, col1 = col0 + 16;
    size_t o0 = (size_t)row * Nn + col0, o1 = (size_t)row * Nn + col1;
    float x0 = c0[v] + bias[col0], x1 = c1[v] + bias[col1];
    if (MODE == EP_ADD) { x0 += C[o0]; x1 += C[o1]; }
    else if (MODE == EP_GATE) {
      x0 = g * x0 + (1.0f - g) * R[o0];
      x1 = g * x1 + (1.0f - g) * R[o1];
    }
    C[o0] = x0; C[o1] = x1;
  }
}

// ---------------------------------------------------------------------------
// Per-head 64x64 GEMM: OUT[n, h*64+:] (op)= IN[n, h*64+:] @ M[h] (opt. M^T)
// grid = (Nrows/64, NH). Used for Q' = Q @ a_rel^T and agg += Wagg @ m_rel.
// ---------------------------------------------------------------------------
template <int TRB, int ACC>
__global__ __launch_bounds__(256)
void headgemm_bf16_wmma(const float* __restrict__ IN, const float* __restrict__ Mt,
                        float* OUT)
{
  __shared__ __bf16 As[2][64][40];   // [m][k]
  __shared__ __bf16 Bt[2][64][40];   // [n][k]
  const int tid  = threadIdx.x;
  const int wave = tid >> 5, lane = tid & 31;
  const int h  = blockIdx.y;
  const float* Mh = Mt + (size_t)h * DH * DH;
  const int mb = blockIdx.x * 64;
  const int wm = (wave >> 1) * 16, wn = (wave & 1) * 32;
  const int m = lane & 15, sel = lane >> 4;
  v8f c0 = {}, c1 = {};
  auto stage = [&](int k0, int p) {
    for (int i = tid; i < 512; i += 256) {
      int r = i >> 3, c4 = (i & 7) * 4;
      float4 f = *(const float4*)&IN[(size_t)(mb + r) * HD + h * DH + k0 + c4];
      v4bf pk = { (__bf16)f.x, (__bf16)f.y, (__bf16)f.z, (__bf16)f.w };
      *(v4bf*)&As[p][r][c4] = pk;
    }
    for (int i = tid; i < 2048; i += 256) {
      int r = i >> 6, c = i & 63;   // r = k row (32), c = n (64)
      float w = TRB ? Mh[(size_t)c * DH + (k0 + r)]
                    : Mh[(size_t)(k0 + r) * DH + c];
      Bt[p][c][r] = (__bf16)w;
    }
  };
  stage(0, 0);
  __syncthreads();
#pragma unroll
  for (int it = 0; it < 2; ++it) {
    const int p = it & 1;
    if (it + 1 < 2) stage(32, p ^ 1);
    const v8bf* ar  = (const v8bf*)&As[p][wm + m][0];
    const v8bf* br0 = (const v8bf*)&Bt[p][wn + m][0];
    const v8bf* br1 = (const v8bf*)&Bt[p][wn + 16 + m][0];
    v16bf a  = cat8(ar[2 * sel], ar[2 * sel + 1]);
    v16bf b0 = cat8(br0[2 * sel], br0[2 * sel + 1]);
    v16bf b1 = cat8(br1[2 * sel], br1[2 * sel + 1]);
    c0 = __builtin_amdgcn_wmma_f32_16x16x32_bf16(false, a, false, b0, (short)0, c0, false, false);
    c1 = __builtin_amdgcn_wmma_f32_16x16x32_bf16(false, a, false, b1, (short)0, c1, false, false);
    __syncthreads();
  }
#pragma unroll
  for (int v = 0; v < 8; ++v) {
    int row = mb + wm + v + 8 * sel;
    size_t o0 = (size_t)row * HD + h * DH + wn + m;
    size_t o1 = o0 + 16;
    float x0 = c0[v], x1 = c1[v];
    if (ACC) { x0 += OUT[o0]; x1 += OUT[o1]; }
    OUT[o0] = x0; OUT[o1] = x1;
  }
}

// ---------------------------------------------------------------------------
// Flash cross-attention: O[q, h*64+:] = softmax(Q K^T * s) V, per head.
// grid = (Nq/64, NH). 64 q-rows per block, k-tiles of 64, online softmax.
// LDS layouts chosen so every fragment is 2x ds_load_b128:
//   Qs[q][d], Ps[q][k] (A-side, contraction contiguous)
//   Kst[k][d] natural  (B of QK^T: contraction d contiguous per k-col row)
//   Vst[d][k] transposed (B of PV: contraction k contiguous per d-col row)
// ---------------------------------------------------------------------------
__global__ __launch_bounds__(256)
void flash_cross_attn(const float* __restrict__ Qb, const float* __restrict__ Kb,
                      const float* __restrict__ Vb, float* __restrict__ Ob,
                      int Nq, int Nk)
{
  __shared__ __bf16 Qs[64][72];
  __shared__ __bf16 Kst[64][72];
  __shared__ __bf16 Vst[64][72];
  __shared__ __bf16 Ps[64][72];
  const int tid  = threadIdx.x;
  const int wave = tid >> 5, lane = tid & 31;
  const int h  = blockIdx.y;
  const int qb = blockIdx.x * 64;
  const int wm = (wave >> 1) * 16, wn = (wave & 1) * 32;
  const int m = lane & 15, sel = lane >> 4;

  for (int i = tid; i < 1024; i += 256) {
    int r = i >> 4, c4 = (i & 15) * 4;
    float4 f = *(const float4*)&Qb[(size_t)(qb + r) * HD + h * DH + c4];
    v4bf p = { (__bf16)(f.x * SCALE), (__bf16)(f.y * SCALE),
               (__bf16)(f.z * SCALE), (__bf16)(f.w * SCALE) };
    *(v4bf*)&Qs[r][c4] = p;
  }
  __syncthreads();

  v16bf qf[2];
  {
    const v8bf* qr = (const v8bf*)&Qs[wm + m][0];
#pragma unroll
    for (int s = 0; s < 2; ++s) qf[s] = cat8(qr[s * 4 + 2 * sel], qr[s * 4 + 2 * sel + 1]);
  }

  v8f o0 = {}, o1 = {};
  float mrun[8], lrun[8];
#pragma unroll
  for (int v = 0; v < 8; ++v) { mrun[v] = -3.0e38f; lrun[v] = 0.0f; }

  for (int kt = 0; kt < Nk; kt += 64) {
    __syncthreads();  // protect Kst/Vst/Ps from previous iteration
    for (int i = tid; i < 1024; i += 256) {
      int r = i >> 4, c4 = (i & 15) * 4;
      float4 fk = *(const float4*)&Kb[(size_t)(kt + r) * HD + h * DH + c4];
      v4bf pk = { (__bf16)fk.x, (__bf16)fk.y, (__bf16)fk.z, (__bf16)fk.w };
      *(v4bf*)&Kst[r][c4] = pk;                       // natural [k][d]
      float4 fv = *(const float4*)&Vb[(size_t)(kt + r) * HD + h * DH + c4];
      Vst[c4 + 0][r] = (__bf16)fv.x; Vst[c4 + 1][r] = (__bf16)fv.y;
      Vst[c4 + 2][r] = (__bf16)fv.z; Vst[c4 + 3][r] = (__bf16)fv.w;
    }
    __syncthreads();

    // S = Q K^T : 4 k-segments of 16 cols, contraction over d (2 steps of 32)
    v8f sc[4] = {};
#pragma unroll
    for (int ds = 0; ds < 2; ++ds) {
#pragma unroll
      for (int s = 0; s < 4; ++s) {
        const v8bf* kr = (const v8bf*)&Kst[s * 16 + m][0];
        v16bf bf = cat8(kr[ds * 4 + 2 * sel], kr[ds * 4 + 2 * sel + 1]);
        sc[s] = __builtin_amdgcn_wmma_f32_16x16x32_bf16(false, qf[ds], false, bf,
                                                        (short)0, sc[s], false, false);
      }
    }

    // online softmax update (row reductions over 16 lanes within each half)
    float mnew[8];
#pragma unroll
    for (int v = 0; v < 8; ++v) {
      float mx = fmaxf(fmaxf(sc[0][v], sc[1][v]), fmaxf(sc[2][v], sc[3][v]));
#pragma unroll
      for (int off = 1; off < 16; off <<= 1) mx = fmaxf(mx, __shfl_xor(mx, off, 32));
      mnew[v] = fmaxf(mrun[v], mx);
    }
#pragma unroll
    for (int v = 0; v < 8; ++v) {
      float alpha = __expf(mrun[v] - mnew[v]);
      o0[v] *= alpha; o1[v] *= alpha;
      float rs = 0.0f;
#pragma unroll
      for (int s = 0; s < 4; ++s) {
        float p = __expf(sc[s][v] - mnew[v]);
        sc[s][v] = p; rs += p;
      }
#pragma unroll
      for (int off = 1; off < 16; off <<= 1) rs += __shfl_xor(rs, off, 32);
      lrun[v] = lrun[v] * alpha + rs;
      mrun[v] = mnew[v];
    }

    // stash P (waves 0,2,4,6 cover the 4 distinct q-row groups)
    if ((wave & 1) == 0) {
#pragma unroll
      for (int v = 0; v < 8; ++v) {
        int row = wm + v + 8 * sel;
#pragma unroll
        for (int s = 0; s < 4; ++s) Ps[row][s * 16 + m] = (__bf16)sc[s][v];
      }
    }
    __syncthreads();

    // O += P V : contraction over k (2 steps of 32)
    {
      const v8bf* pr  = (const v8bf*)&Ps[wm + m][0];
      const v8bf* vr0 = (const v8bf*)&Vst[wn + m][0];
      const v8bf* vr1 = (const v8bf*)&Vst[wn + 16 + m][0];
#pragma unroll
      for (int ks = 0; ks < 2; ++ks) {
        v16bf pa  = cat8(pr[ks * 4 + 2 * sel], pr[ks * 4 + 2 * sel + 1]);
        v16bf vb0 = cat8(vr0[ks * 4 + 2 * sel], vr0[ks * 4 + 2 * sel + 1]);
        v16bf vb1 = cat8(vr1[ks * 4 + 2 * sel], vr1[ks * 4 + 2 * sel + 1]);
        o0 = __builtin_amdgcn_wmma_f32_16x16x32_bf16(false, pa, false, vb0, (short)0, o0, false, false);
        o1 = __builtin_amdgcn_wmma_f32_16x16x32_bf16(false, pa, false, vb1, (short)0, o1, false, false);
      }
    }
  }

#pragma unroll
  for (int v = 0; v < 8; ++v) {
    int row = qb + wm + v + 8 * sel;
    float inv = 1.0f / lrun[v];
    Ob[(size_t)row * HD + h * DH + wn + m]      = o0[v] * inv;
    Ob[(size_t)row * HD + h * DH + wn + 16 + m] = o1[v] * inv;
  }
}

// ---------------------------------------------------------------------------
// Edge kernels (memory-bound phase, ~2-3 GB/layer at 23.3 TB/s)
// ---------------------------------------------------------------------------
__global__ void edge_logits(const float* __restrict__ Qp, const float* __restrict__ Kt,
                            const int* __restrict__ src, const int* __restrict__ dst,
                            const float* __restrict__ prel, float* __restrict__ alog,
                            float* amax, int E)
{
  int idx = blockIdx.x * blockDim.x + threadIdx.x;
  int e = idx >> 2, h = idx & 3;
  if (e >= E) return;
  int s = src[e], d = dst[e];
  const float4* qv = (const float4*)(Qp + (size_t)d * HD + h * DH);
  const float4* kv = (const float4*)(Kt + (size_t)s * HD + h * DH);
  float acc = 0.0f;
#pragma unroll
  for (int i = 0; i < 16; ++i) {
    float4 a = qv[i], b = kv[i];
    acc += a.x * b.x + a.y * b.y + a.z * b.z + a.w * b.w;
  }
  acc *= prel[h] * SCALE;
  alog[(size_t)e * NH + h] = acc;
  atomic_max_f(&amax[d * NH + h], acc);
}

__global__ void edge_exp(const int* __restrict__ dst, const float* __restrict__ amax,
                         float* alog, float* den, int E)
{
  int idx = blockIdx.x * blockDim.x + threadIdx.x;
  int e = idx >> 2, h = idx & 3;
  if (e >= E) return;
  int d = dst[e];
  float ex = __expf(alog[(size_t)e * NH + h] - amax[d * NH + h]);
  alog[(size_t)e * NH + h] = ex;
  atomicAdd(&den[d * NH + h], ex);
}

__global__ void edge_scatter(const int* __restrict__ src, const int* __restrict__ dst,
                             const float* __restrict__ ex, const float* __restrict__ den,
                             const float* __restrict__ V, float* Wagg, int E)
{
  int idx = blockIdx.x * blockDim.x + threadIdx.x;
  int e = idx >> 6, l = idx & 63;
  if (e >= E) return;
  int h = l >> 4;  // 16 float4-lanes per head
  int s = src[e], d = dst[e];
  float w = ex[(size_t)e * NH + h] / den[d * NH + h];
  float4 v = *(const float4*)(V + (size_t)s * HD + l * 4);
  float* o = Wagg + (size_t)d * HD + l * 4;
  atomicAdd(o + 0, w * v.x); atomicAdd(o + 1, w * v.y);
  atomicAdd(o + 2, w * v.z); atomicAdd(o + 3, w * v.w);
}

// ---------------------------------------------------------------------------
// Fused residual + LayerNorm + exact GELU. One wave per row (8 rows / block).
// ---------------------------------------------------------------------------
__global__ __launch_bounds__(256)
void ln_gelu(const float* hin, const float* hadd, const float* __restrict__ gam,
             const float* __restrict__ bet, float* out, int Nrows)
{
  int wave = threadIdx.x >> 5, lane = threadIdx.x & 31;
  int row = blockIdx.x * 8 + wave;
  if (row >= Nrows) return;
  float x[8];
  float s = 0.0f;
#pragma unroll
  for (int i = 0; i < 8; ++i) {
    int c = i * 32 + lane;
    x[i] = hin[(size_t)row * HD + c] + hadd[(size_t)row * HD + c];
    s += x[i];
  }
#pragma unroll
  for (int off = 1; off < 32; off <<= 1) s += __shfl_xor(s, off, 32);
  float mean = s * (1.0f / HD);
  float vs = 0.0f;
#pragma unroll
  for (int i = 0; i < 8; ++i) { float d = x[i] - mean; vs += d * d; }
#pragma unroll
  for (int off = 1; off < 32; off <<= 1) vs += __shfl_xor(vs, off, 32);
  float r = rsqrtf(vs * (1.0f / HD) + 1e-5f);
#pragma unroll
  for (int i = 0; i < 8; ++i) {
    int c = i * 32 + lane;
    float y = (x[i] - mean) * r * gam[c] + bet[c];
    out[(size_t)row * HD + c] = gelu_f(y);
  }
}

// ---------------------------------------------------------------------------
// Host orchestration
// ---------------------------------------------------------------------------
extern "C" void kernel_launch(void* const* d_in, const int* in_sizes, int n_in,
                              void* d_out, int out_size, void* d_ws, size_t ws_size,
                              hipStream_t stream)
{
  (void)in_sizes; (void)n_in; (void)out_size; (void)ws_size;
  const float* x_p = (const float*)d_in[0];
  const float* x_g = (const float*)d_in[1];
  const float* x_m = (const float*)d_in[2];
  const int* srcR[4] = {(const int*)d_in[3], (const int*)d_in[5],
                        (const int*)d_in[7], (const int*)d_in[9]};
  const int* dstR[4] = {(const int*)d_in[4], (const int*)d_in[6],
                        (const int*)d_in[8], (const int*)d_in[10]};
  const float* Win[3]  = {(const float*)d_in[11], (const float*)d_in[13], (const float*)d_in[15]};
  const float* binp[3] = {(const float*)d_in[12], (const float*)d_in[14], (const float*)d_in[16]};
  const float* kw = (const float*)d_in[17];
  const float* qw = (const float*)d_in[18];
  const float* vw = (const float*)d_in[19];
  const float* aw = (const float*)d_in[20];
  const float* kb = (const float*)d_in[21];
  const float* qb = (const float*)d_in[22];
  const float* vb = (const float*)d_in[23];
  const float* ab = (const float*)d_in[24];
  const float* skip  = (const float*)d_in[25];
  const float* a_rel = (const float*)d_in[26];
  const float* m_rel = (const float*)d_in[27];
  const float* p_rel = (const float*)d_in[28];
  const float* ca_qw = (const float*)d_in[29];
  const float* ca_kw = (const float*)d_in[30];
  const float* ca_vw = (const float*)d_in[31];
  const float* ca_ow = (const float*)d_in[32];
  const float* ca_qb = (const float*)d_in[33];
  const float* ca_kb = (const float*)d_in[34];
  const float* ca_vb = (const float*)d_in[35];
  const float* ca_ob = (const float*)d_in[36];
  const float* lng = (const float*)d_in[37];
  const float* lnb = (const float*)d_in[38];
  const float* outw = (const float*)d_in[39];
  const float* outb = (const float*)d_in[40];
  float* out = (float*)d_out;

  const int Ns[3]   = {2048, 4096, 3072};
  const int IN_D[3] = {512, 768, 384};
  const int tp[3]   = {0, 2048, 6144};
  const int Er[4]   = {250000, 250000, 200000, 200000};
  const int rs[4]   = {0, 1, 0, 2};
  const int rd[4]   = {1, 0, 2, 0};
  const int ep[4]   = {0, 250000, 500000, 700000};
  const int wp[4]   = {0, 4096, 6144, 9216};  // wagg row prefixes (4096,2048,3072,2048)
  const int NT = 9216;

  float* base = (float*)d_ws;
  size_t off = 0;
  auto alloc = [&](size_t n) { float* p = base + off; off += n; return p; };
  float* hT     = alloc((size_t)NT * HD);
  float* hgT    = alloc((size_t)NT * HD);
  float* KbT    = alloc((size_t)NT * HD);
  float* QbT    = alloc((size_t)NT * HD);
  float* VbT    = alloc((size_t)NT * HD);
  float* aggT   = alloc((size_t)NT * HD);
  float* qprime = alloc((size_t)4096 * HD);
  float* wagg   = alloc((size_t)11264 * HD);
  float* alog   = alloc((size_t)900000 * NH);
  float* amax   = alloc((size_t)NT * NH);
  float* den    = alloc((size_t)NT * NH);
  float* caq    = alloc((size_t)4096 * HD);
  float* cak    = alloc((size_t)2048 * HD);
  float* cav    = alloc((size_t)2048 * HD);
  float* cao    = alloc((size_t)4096 * HD);

  float *h[3], *hg[3], *Kt[3], *Qt[3], *Vt[3], *agg[3], *amaxT[3], *denT[3];
  for (int t = 0; t < 3; ++t) {
    h[t]  = hT  + (size_t)tp[t] * HD;   hg[t]  = hgT  + (size_t)tp[t] * HD;
    Kt[t] = KbT + (size_t)tp[t] * HD;   Qt[t]  = QbT  + (size_t)tp[t] * HD;
    Vt[t] = VbT + (size_t)tp[t] * HD;   agg[t] = aggT + (size_t)tp[t] * HD;
    amaxT[t] = amax + (size_t)tp[t] * NH;
    denT[t]  = den  + (size_t)tp[t] * NH;
  }

  // ---- input projections: h[t] = x_t @ Win_t + bin_t
  {
    const float* xs[3] = {x_p, x_g, x_m};
    for (int t = 0; t < 3; ++t) {
      dim3 g(Ns[t] / 64, HD / 64);
      gemm_bf16_wmma<EP_BIAS, 0><<<g, 256, 0, stream>>>(xs[t], Win[t], binp[t], h[t],
                                                        nullptr, nullptr, Ns[t], IN_D[t], HD);
    }
  }

  for (int l = 0; l < 2; ++l) {
    // ---- per-type K/Q/V projections
    for (int t = 0; t < 3; ++t) {
      dim3 g(Ns[t] / 64, HD / 64);
      size_t wo = (size_t)(l * 3 + t) * HD * HD;
      size_t bo = (size_t)(l * 3 + t) * HD;
      gemm_bf16_wmma<EP_BIAS, 0><<<g, 256, 0, stream>>>(h[t], kw + wo, kb + bo, Kt[t], nullptr, nullptr, Ns[t], HD, HD);
      gemm_bf16_wmma<EP_BIAS, 0><<<g, 256, 0, stream>>>(h[t], qw + wo, qb + bo, Qt[t], nullptr, nullptr, Ns[t], HD, HD);
      gemm_bf16_wmma<EP_BIAS, 0><<<g, 256, 0, stream>>>(h[t], vw + wo, vb + bo, Vt[t], nullptr, nullptr, Ns[t], HD, HD);
    }
    // ---- reset softmax / aggregation state (0xFF floats work with the
    //      sign-split atomic max: int -1 / uint max)
    hipMemsetAsync(amax, 0xFF, (size_t)NT * NH * sizeof(float), stream);
    hipMemsetAsync(den, 0, (size_t)NT * NH * sizeof(float), stream);
    hipMemsetAsync(wagg, 0, (size_t)11264 * HD * sizeof(float), stream);
    hipMemsetAsync(aggT, 0, (size_t)NT * HD * sizeof(float), stream);

    // ---- logits: a = (Q @ a_rel^T)[dst] . K[src] * p_rel * SCALE; seg-max
    for (int r = 0; r < 4; ++r) {
      int t = rd[r], s = rs[r];
      dim3 g(Ns[t] / 64, NH);
      headgemm_bf16_wmma<1, 0><<<g, 256, 0, stream>>>(Qt[t], a_rel + (size_t)(l * 4 + r) * NH * DH * DH,
                                                      qprime);
      int blk = (Er[r] * NH + 255) / 256;
      edge_logits<<<blk, 256, 0, stream>>>(qprime, Kt[s], srcR[r], dstR[r],
                                           p_rel + (size_t)(l * 4 + r) * NH,
                                           alog + (size_t)ep[r] * NH, amaxT[t], Er[r]);
    }
    // ---- exp + segment sum
    for (int r = 0; r < 4; ++r) {
      int blk = (Er[r] * NH + 255) / 256;
      edge_exp<<<blk, 256, 0, stream>>>(dstR[r], amaxT[rd[r]],
                                        alog + (size_t)ep[r] * NH, denT[rd[r]], Er[r]);
    }
    // ---- weighted scatter of raw V rows into per-relation accumulators
    for (int r = 0; r < 4; ++r) {
      int blk = (Er[r] * 64 + 255) / 256;
      edge_scatter<<<blk, 256, 0, stream>>>(srcR[r], dstR[r], alog + (size_t)ep[r] * NH,
                                            denT[rd[r]], Vt[rs[r]],
                                            wagg + (size_t)wp[r] * HD, Er[r]);
    }
    // ---- agg[t] += Wagg_r @ m_rel[r]  (hoisted message transform)
    for (int r = 0; r < 4; ++r) {
      int t = rd[r];
      dim3 g(Ns[t] / 64, NH);
      headgemm_bf16_wmma<0, 1><<<g, 256, 0, stream>>>(wagg + (size_t)wp[r] * HD,
                                                      m_rel + (size_t)(l * 4 + r) * NH * DH * DH,
                                                      agg[t]);
    }
    // ---- hg[t] = sigmoid(skip)*(gelu(agg)@aw+ab) + (1-sigmoid(skip))*h[t]
    for (int t = 0; t < 3; ++t) {
      dim3 g(Ns[t] / 64, HD / 64);
      size_t wo = (size_t)(l * 3 + t) * HD * HD;
      gemm_bf16_wmma<EP_GATE, 1><<<g, 256, 0, stream>>>(agg[t], aw + wo, ab + (size_t)(l * 3 + t) * HD,
                                                        hg[t], h[t], skip + (l * 3 + t),
                                                        Ns[t], HD, HD);
    }
    // ---- cross attention: genes/metabolites attend to participants
    for (int c = 0; c < 2; ++c) {
      int t = c + 1;
      dim3 gq(Ns[t] / 64, HD / 64), gk(Ns[0] / 64, HD / 64);
      gemm_bf16_wmma<EP_BIAS, 0><<<gq, 256, 0, stream>>>(h[t], ca_qw + (size_t)c * HD * HD, ca_qb + c * HD,
                                                         caq, nullptr, nullptr, Ns[t], HD, HD);
      gemm_bf16_wmma<EP_BIAS, 0><<<gk, 256, 0, stream>>>(h[0], ca_kw + (size_t)c * HD * HD, ca_kb + c * HD,
                                                         cak, nullptr, nullptr, Ns[0], HD, HD);
      gemm_bf16_wmma<EP_BIAS, 0><<<gk, 256, 0, stream>>>(h[0], ca_vw + (size_t)c * HD * HD, ca_vb + c * HD,
                                                         cav, nullptr, nullptr, Ns[0], HD, HD);
      dim3 gf(Ns[t] / 64, NH);
      flash_cross_attn<<<gf, 256, 0, stream>>>(caq, cak, cav, cao, Ns[t], Ns[0]);
      gemm_bf16_wmma<EP_ADD, 0><<<gq, 256, 0, stream>>>(cao, ca_ow + (size_t)c * HD * HD, ca_ob + c * HD,
                                                        hg[t], nullptr, nullptr, Ns[t], HD, HD);
    }
    // ---- h[t] = gelu(LN(h[t] + hg[t]))
    for (int t = 0; t < 3; ++t) {
      ln_gelu<<<Ns[t] / 8, 256, 0, stream>>>(h[t], hg[t], lng + t * HD, lnb + t * HD, h[t], Ns[t]);
    }
  }

  // ---- output projections, concatenated [participant | gene | metabolite]
  {
    const int op[3] = {0, 2048 * OD, (2048 + 4096) * OD};
    for (int t = 0; t < 3; ++t) {
      dim3 g(Ns[t] / 64, OD / 64);
      gemm_bf16_wmma<EP_BIAS, 0><<<g, 256, 0, stream>>>(h[t], outw + (size_t)t * HD * OD, outb + t * OD,
                                                        out + op[t], nullptr, nullptr,
                                                        Ns[t], HD, OD);
    }
  }
}
